// SimpleAttention_4294967296116
// MI455X (gfx1250) — matrix operand order repaired and checked by reading, not compiled, not run on hardware
//
#include <hip/hip_runtime.h>
#include <stdint.h>

typedef __attribute__((ext_vector_type(16))) __bf16       v16bf;
typedef __attribute__((ext_vector_type(8)))  float        v8f;
typedef __attribute__((ext_vector_type(4)))  unsigned int v4u;
typedef __attribute__((ext_vector_type(8)))  unsigned int v8u;

#define SB 4
#define SS 4096
#define SE 1024

__device__ __forceinline__ uint16_t f2bf(float f) {
  union { float f; uint32_t u; } c; c.f = f;
  uint32_t u = c.u;
  uint32_t r = u + 0x7FFFu + ((u >> 16) & 1u);   // round-to-nearest-even
  return (uint16_t)(r >> 16);
}

// -------- elementwise fp32 -> bf16 (vectorized x4) --------
__global__ void __launch_bounds__(256)
cvt_f32_bf16_x4(const float* __restrict__ in, uint16_t* __restrict__ out, long long n4) {
  long long i = (long long)blockIdx.x * blockDim.x + threadIdx.x;
  if (i >= n4) return;
  float4 f = reinterpret_cast<const float4*>(in)[i];
  ushort4 o;
  o.x = f2bf(f.x); o.y = f2bf(f.y); o.z = f2bf(f.z); o.w = f2bf(f.w);
  reinterpret_cast<ushort4*>(out)[i] = o;
}

// -------- 1024x1024 weight transpose + convert: Wt[e][f] = bf16(W[f][e]) --------
__global__ void __launch_bounds__(256)
wt_transpose_bf16(const float* __restrict__ in, uint16_t* __restrict__ out) {
  int f = blockIdx.x * 32 + (threadIdx.x & 31);
  int e = blockIdx.y * 8 + (threadIdx.x >> 5);
  out[(uint32_t)e * SE + f] = f2bf(in[(uint32_t)f * SE + e]);
}

// -------- fragment load: two 16B chunks (base + 32-bit elem offsets) --------
__device__ __forceinline__ v16bf ld_frag(const uint16_t* __restrict__ base,
                                         uint32_t off0, uint32_t off1) {
  v4u lo = *reinterpret_cast<const v4u*>(base + off0);
  v4u hi = *reinterpret_cast<const v4u*>(base + off1);
  v8u t = __builtin_shufflevector(lo, hi, 0, 1, 2, 3, 4, 5, 6, 7);
  return __builtin_bit_cast(v16bf, t);
}

// ---------------------------------------------------------------------------
// Generic batched GEMM: C[M,N] = A[M,K] (row-major bf16) * B[K,N] (K-major
// bf16) + bias[N].  Block = 128 threads = 4 waves; block tile 64x128; wave
// tile 32x64 (2x4 accumulators of v_wmma_f32_16x16x32_bf16).
// Addressing: uniform SGPR base + 32-bit per-lane element offsets so loads
// lower to global_load_b128 (GVS mode), not flat_load.
// mode: 0 = store bf16 row-major, 1 = store bf16 transposed ([N][M]),
//       2 = store f32 row-major.
// ---------------------------------------------------------------------------
__global__ void __launch_bounds__(128)
gemm_bf16_wmma(const uint16_t* __restrict__ A, int lda, long long strideA,
               const uint16_t* __restrict__ B, int ldb, long long strideB,
               const float* __restrict__ bias,
               void* __restrict__ Cv, int ldc, long long strideC,
               int K, int mode) {
  const int lane = threadIdx.x & 31;
  const int wave = threadIdx.x >> 5;
  const int wm = wave >> 1;
  const int wn = wave & 1;
  const uint32_t mWave = blockIdx.y * 64u + wm * 32u;
  const uint32_t nWave = blockIdx.x * 128u + wn * 64u;

  const uint16_t* __restrict__ Ab = A + (long long)blockIdx.z * strideA;
  const uint16_t* __restrict__ Bb = B + (long long)blockIdx.z * strideB;

  const uint32_t hi16 = lane >> 4;   // which half-wave
  const uint32_t l15  = lane & 15;

  // A frag (16x32): lane = M row; per-lane chunks at K = hi16*8 and hi16*8+16
  uint32_t aOff[2];
#pragma unroll
  for (int i = 0; i < 2; ++i)
    aOff[i] = (mWave + i * 16u + l15) * (uint32_t)lda + hi16 * 8u;

  // B frag (32x16): lane&15 = N column; per-lane K = hi16*8+s (s<8), hi16*8+16+(s-8) (s>=8)
  uint32_t bOff[4];
#pragma unroll
  for (int j = 0; j < 4; ++j)
    bOff[j] = hi16 * 8u * (uint32_t)ldb + nWave + j * 16u + l15;

  const v8f zero = {0.f, 0.f, 0.f, 0.f, 0.f, 0.f, 0.f, 0.f};
  v8f acc[2][4];
#pragma unroll
  for (int i = 0; i < 2; ++i)
#pragma unroll
    for (int j = 0; j < 4; ++j) acc[i][j] = zero;

  const uint32_t bStep = (uint32_t)ldb * 32u;
  for (int k0 = 0; k0 < K; k0 += 32) {
    v16bf af[2], bfr[4];
#pragma unroll
    for (int i = 0; i < 2; ++i) af[i] = ld_frag(Ab, aOff[i], aOff[i] + 16u);
#pragma unroll
    for (int j = 0; j < 4; ++j) {
      v16bf t;
#pragma unroll
      for (int s = 0; s < 16; ++s)
        t[s] = __builtin_bit_cast(__bf16, Bb[bOff[j] + (uint32_t)((s & 7) + ((s >> 3) << 4)) * (uint32_t)ldb]);
      bfr[j] = t;
    }
#pragma unroll
    for (int i = 0; i < 2; ++i)
#pragma unroll
      for (int j = 0; j < 4; ++j)
        acc[i][j] = __builtin_amdgcn_wmma_f32_16x16x32_bf16(
            false, af[i], false, bfr[j], (short)0, acc[i][j], false, false);
#pragma unroll
    for (int i = 0; i < 2; ++i) aOff[i] += 32u;
#pragma unroll
    for (int j = 0; j < 4; ++j) bOff[j] += bStep;
  }

  float bb[4] = {0.f, 0.f, 0.f, 0.f};
  if (bias) {
#pragma unroll
    for (int j = 0; j < 4; ++j) bb[j] = bias[nWave + j * 16u + l15];
  }

  // C layout: VGPR r -> M = r + 8*hi16 (per 16-row tile), lane&15 -> N
  if (mode == 0) {
    uint16_t* __restrict__ Cp =
        reinterpret_cast<uint16_t*>(Cv) + (long long)blockIdx.z * strideC;
#pragma unroll
    for (int i = 0; i < 2; ++i)
#pragma unroll
      for (int j = 0; j < 4; ++j)
#pragma unroll
        for (int r = 0; r < 8; ++r) {
          uint32_t m = mWave + i * 16u + r + 8u * hi16;
          uint32_t n = nWave + j * 16u + l15;
          Cp[m * (uint32_t)ldc + n] = f2bf(acc[i][j][r] + bb[j]);
        }
  } else if (mode == 1) {
    uint16_t* __restrict__ Cp =
        reinterpret_cast<uint16_t*>(Cv) + (long long)blockIdx.z * strideC;
#pragma unroll
    for (int i = 0; i < 2; ++i)
#pragma unroll
      for (int j = 0; j < 4; ++j)
#pragma unroll
        for (int r = 0; r < 8; ++r) {
          uint32_t m = mWave + i * 16u + r + 8u * hi16;
          uint32_t n = nWave + j * 16u + l15;
          Cp[n * (uint32_t)ldc + m] = f2bf(acc[i][j][r] + bb[j]);
        }
  } else {
    float* __restrict__ Cp =
        reinterpret_cast<float*>(Cv) + (long long)blockIdx.z * strideC;
#pragma unroll
    for (int i = 0; i < 2; ++i)
#pragma unroll
      for (int j = 0; j < 4; ++j)
#pragma unroll
        for (int r = 0; r < 8; ++r) {
          uint32_t m = mWave + i * 16u + r + 8u * hi16;
          uint32_t n = nWave + j * 16u + l15;
          Cp[m * (uint32_t)ldc + n] = acc[i][j][r] + bb[j];
        }
  }
}

extern "C" void kernel_launch(void* const* d_in, const int* in_sizes, int n_in,
                              void* d_out, int out_size, void* d_ws, size_t ws_size,
                              hipStream_t stream) {
  (void)in_sizes; (void)n_in; (void)out_size; (void)ws_size;
  const float* x  = (const float*)d_in[0];
  const float* Wq = (const float*)d_in[1];
  const float* bq = (const float*)d_in[2];
  const float* Wk = (const float*)d_in[3];
  const float* bk = (const float*)d_in[4];
  const float* Wv = (const float*)d_in[5];
  const float* bv = (const float*)d_in[6];
  const float* Wo = (const float*)d_in[7];
  const float* bo = (const float*)d_in[8];
  float* out = (float*)d_out;

  const long long M  = (long long)SB * SS;   // 16384 rows total
  const long long xN = M * SE;               // 16,777,216 elements

  char* ws = (char*)d_ws;
  uint16_t* xb  = (uint16_t*)ws; ws += xN * 2;
  uint16_t* Wqt = (uint16_t*)ws; ws += (long long)SE * SE * 2;
  uint16_t* Wkt = (uint16_t*)ws; ws += (long long)SE * SE * 2;
  uint16_t* Wvt = (uint16_t*)ws; ws += (long long)SE * SE * 2;
  uint16_t* Wot = (uint16_t*)ws; ws += (long long)SE * SE * 2;
  uint16_t* qb  = (uint16_t*)ws; ws += xN * 2;
  uint16_t* kTb = (uint16_t*)ws; ws += xN * 2;              // per batch [E][S]
  uint16_t* vb  = (uint16_t*)ws; ws += xN * 2;
  uint16_t* scb = (uint16_t*)ws; ws += (long long)SB * SS * SS * 2;
  uint16_t* atb = (uint16_t*)ws; ws += xN * 2;

  // ---- convert inputs to bf16 ----
  cvt_f32_bf16_x4<<<dim3((unsigned)(xN / 4 / 256)), dim3(256), 0, stream>>>(x, xb, xN / 4);
  dim3 tg(32, 128);
  wt_transpose_bf16<<<tg, 256, 0, stream>>>(Wq, Wqt);
  wt_transpose_bf16<<<tg, 256, 0, stream>>>(Wk, Wkt);
  wt_transpose_bf16<<<tg, 256, 0, stream>>>(Wv, Wvt);
  wt_transpose_bf16<<<tg, 256, 0, stream>>>(Wo, Wot);

  dim3 blk(128);
  const long long sSE = (long long)SS * SE;  // 4096*1024
  const long long sES = (long long)SE * SS;
  const long long sSS = (long long)SS * SS;

  // q = x @ Wq^T + bq   -> bf16 row-major [B*S, E]
  gemm_bf16_wmma<<<dim3(SE / 128, (unsigned)(M / 64), 1), blk, 0, stream>>>(
      xb, SE, 0, Wqt, SE, 0, bq, qb, SE, 0, SE, 0);
  // k = x @ Wk^T + bk   -> stored TRANSPOSED per batch: kT[b][e][s]
  gemm_bf16_wmma<<<dim3(SE / 128, SS / 64, SB), blk, 0, stream>>>(
      xb, SE, sSE, Wkt, SE, 0, bk, kTb, SS, sES, SE, 1);
  // v = x @ Wv^T + bv   -> bf16 row-major
  gemm_bf16_wmma<<<dim3(SE / 128, (unsigned)(M / 64), 1), blk, 0, stream>>>(
      xb, SE, 0, Wvt, SE, 0, bv, vb, SE, 0, SE, 0);
  // scores[b] = q[b] @ kT[b]   [S,S], K = E
  gemm_bf16_wmma<<<dim3(SS / 128, SS / 64, SB), blk, 0, stream>>>(
      qb, SE, sSE, kTb, SS, sES, nullptr, scb, SS, sSS, SE, 0);
  // attn[b] = scores[b] @ v[b]  [S,E], K = S  (v rows are K-major already)
  gemm_bf16_wmma<<<dim3(SE / 128, SS / 64, SB), blk, 0, stream>>>(
      scb, SS, sSS, vb, SE, sSE, nullptr, atb, SE, sSE, SS, 0);
  // out = attn @ Wo^T + bo  -> f32 to d_out
  gemm_bf16_wmma<<<dim3(SE / 128, (unsigned)(M / 64), 1), blk, 0, stream>>>(
      atb, SE, 0, Wot, SE, 0, bo, out, SE, 0, SE, 2);
}
